// Model_12034498363580
// MI455X (gfx1250) — matrix operand order repaired and checked by reading, not compiled, not run on hardware
//
#include <hip/hip_runtime.h>

typedef __attribute__((ext_vector_type(16))) __bf16 v16bf;
typedef __attribute__((ext_vector_type(8)))  __bf16 v8bf;
typedef __attribute__((ext_vector_type(8)))  float  v8f;

#define WMMA_BF16(a,b,c) \
  __builtin_amdgcn_wmma_f32_16x16x32_bf16(false,(a),false,(b),(short)0,(c),false,false)

__device__ __forceinline__ float sigm(float x){ return 1.0f/(1.0f + __expf(-x)); }

// ---- A-fragment (16x32 bf16): k = 8*half + j (j<8), 16+8*half+(j-8) (j>=8) ----
__device__ __forceinline__ v16bf load_afrag(const __bf16* p, int kh){
  v8bf lo = *reinterpret_cast<const v8bf*>(p + 8*kh);
  v8bf hi = *reinterpret_cast<const v8bf*>(p + 16 + 8*kh);
  v16bf r;
#pragma unroll
  for (int j=0;j<8;++j){ r[j]=lo[j]; r[8+j]=hi[j]; }
  return r;
}
// ---- B-fragment (32x16 bf16), row ptr = W[n] + kt: k = 8*half + j (j<8), 16+8*half+(j-8) (j>=8) ----
__device__ __forceinline__ v16bf load_bfragB(const __bf16* p, int kh){
  v8bf lo = *reinterpret_cast<const v8bf*>(p + 8*kh);
  v8bf hi = *reinterpret_cast<const v8bf*>(p + 16 + 8*kh);
  v16bf r;
#pragma unroll
  for (int j=0;j<8;++j){ r[j]=lo[j]; r[8+j]=hi[j]; }
  return r;
}

// =====================================================================
// Phase 1: bidirectional LSTM.  grid=(64,2), block=512 (16 waves).
// Each WG: 16 sequences; wave w owns h-cols [w*32, w*32+32) across all 4
// gates (8 accumulator tiles = 64 VGPRs) so the c/h update never leaves
// registers.  A zero offset laundered through empty asm (renewed every time
// step) stops LICM from hoisting the t-invariant weight loads (which caused
// scratch spilling), while keeping the base pointer's global address-space
// provenance so the loads stay global_load_b128 (not flat_load, which would
// couple LOADcnt with DScnt and burn LDS bandwidth).
// =====================================================================
#define HDIM 512
#define EPAD 320

__global__ __launch_bounds__(512) void lstm_kernel(
    const __bf16* __restrict__ xbf, const __bf16* __restrict__ Wih,
    const __bf16* __restrict__ Whh, const float* __restrict__ bsum,
    float* __restrict__ uttb)
{
  __shared__ __align__(16) __bf16 h_lds[16*HDIM];
  const int tid  = threadIdx.x;
  const int wave = tid >> 5, lane = tid & 31;
  const int kh   = lane >> 4, lm = lane & 15;
  const int dir  = blockIdx.y;
  const int rowbase = blockIdx.x * 16;

  const __bf16* Wih_d = Wih + (size_t)dir * 2048 * EPAD;
  const __bf16* Whh_d = Whh + (size_t)dir * 2048 * HDIM;
  const float*  bs_d  = bsum + dir * 2048;

  for (int idx = tid; idx < 16*HDIM; idx += 512) h_lds[idx] = (__bf16)0.0f;
  __syncthreads();

  float bias[4][2];
#pragma unroll
  for (int g=0; g<4; ++g)
#pragma unroll
    for (int ct=0; ct<2; ++ct)
      bias[g][ct] = bs_d[g*512 + wave*32 + ct*16 + lm];

  float c_reg[2][8];
#pragma unroll
  for (int ct=0; ct<2; ++ct)
#pragma unroll
    for (int r=0; r<8; ++r) c_reg[ct][r] = 0.0f;

  const __bf16* xrow_base = xbf + (size_t)(rowbase + lm) * 32 * EPAD;

  for (int t = 0; t < 32; ++t){
    const int te = dir ? (31 - t) : t;

    // Opaque zero offset, renewed per step: blocks LICM of the weight loads
    // but preserves global address-space provenance of the bases.
    int woff = 0;
    asm volatile("" : "+s"(woff));
    const __bf16* WihT = Wih_d + woff;
    const __bf16* WhhT = Whh_d + woff;

    v8f acc[4][2];
#pragma unroll
    for (int g=0; g<4; ++g)
#pragma unroll
      for (int ct=0; ct<2; ++ct)
#pragma unroll
        for (int r=0; r<8; ++r) acc[g][ct][r] = bias[g][ct];

    // gates += x_t @ Wih^T   (K = 320, zero padded)
    const __bf16* xr = xrow_base + (size_t)te * EPAD;
    for (int kt = 0; kt < EPAD; kt += 32){
      v16bf a = load_afrag(xr + kt, kh);
#pragma unroll
      for (int g=0; g<4; ++g)
#pragma unroll
        for (int ct=0; ct<2; ++ct){
          const __bf16* br = WihT + (size_t)(g*512 + wave*32 + ct*16 + lm) * EPAD + kt;
          v16bf b = load_bfragB(br, kh);
          acc[g][ct] = WMMA_BF16(a, b, acc[g][ct]);
        }
    }
    // gates += h @ Whh^T   (K = 512, h streamed from LDS)
    for (int kt = 0; kt < HDIM; kt += 32){
      v16bf a = load_afrag(&h_lds[lm*HDIM + kt], kh);
#pragma unroll
      for (int g=0; g<4; ++g)
#pragma unroll
        for (int ct=0; ct<2; ++ct){
          const __bf16* br = WhhT + (size_t)(g*512 + wave*32 + ct*16 + lm) * HDIM + kt;
          v16bf b = load_bfragB(br, kh);
          acc[g][ct] = WMMA_BF16(a, b, acc[g][ct]);
        }
    }
    __syncthreads();   // all reads of previous h done
    // i,f,g,o for the wave's h-slice are all resident -> c update in registers
#pragma unroll
    for (int ct=0; ct<2; ++ct){
      const int hcol = wave*32 + ct*16 + lm;
#pragma unroll
      for (int r=0; r<8; ++r){
        const int m = 8*kh + r;
        float iv = acc[0][ct][r], fv = acc[1][ct][r];
        float gv = acc[2][ct][r], ov = acc[3][ct][r];
        float c  = sigm(fv)*c_reg[ct][r] + sigm(iv)*tanhf(gv);
        c_reg[ct][r] = c;
        float h = sigm(ov)*tanhf(c);
        h_lds[m*HDIM + hcol] = (__bf16)h;
        if (t == 31){
          int n = rowbase + m;               // n = b*64 + u
          uttb[(size_t)((n & 63)*16 + (n >> 6))*1024 + dir*512 + hcol] = h;
        }
      }
    }
    __syncthreads();
  }
}

// =====================================================================
// Phase 2 helpers
// =====================================================================
__device__ __forceinline__ void gemm_strip(
    const float* A0, const float* A1, int lda,
    const __bf16* __restrict__ Bw, int ldb, int K,
    float* out, int ldo, int col0, int lane)
{
  const int kh = lane >> 4, lm = lane & 15;
  v8f acc;
#pragma unroll
  for (int r=0;r<8;++r) acc[r] = 0.0f;
  const float* ar0 = A0 + (size_t)lm * lda;
  const float* ar1 = A1 ? (A1 + (size_t)lm * lda) : nullptr;
  const __bf16* br = Bw + (size_t)(col0 + lm) * ldb;
  for (int kt = 0; kt < K; kt += 32){
    const int kA = kt + 8*kh, kB = kt + 16 + 8*kh;
    float4 l0 = *(const float4*)(ar0 + kA);
    float4 l1 = *(const float4*)(ar0 + kA + 4);
    float4 h0 = *(const float4*)(ar0 + kB);
    float4 h1 = *(const float4*)(ar0 + kB + 4);
    if (ar1){
      float4 m0 = *(const float4*)(ar1 + kA);
      float4 m1 = *(const float4*)(ar1 + kA + 4);
      float4 m2 = *(const float4*)(ar1 + kB);
      float4 m3 = *(const float4*)(ar1 + kB + 4);
      l0.x+=m0.x; l0.y+=m0.y; l0.z+=m0.z; l0.w+=m0.w;
      l1.x+=m1.x; l1.y+=m1.y; l1.z+=m1.z; l1.w+=m1.w;
      h0.x+=m2.x; h0.y+=m2.y; h0.z+=m2.z; h0.w+=m2.w;
      h1.x+=m3.x; h1.y+=m3.y; h1.z+=m3.z; h1.w+=m3.w;
    }
    v16bf a;
    a[0]=(__bf16)l0.x;  a[1]=(__bf16)l0.y;  a[2]=(__bf16)l0.z;  a[3]=(__bf16)l0.w;
    a[4]=(__bf16)l1.x;  a[5]=(__bf16)l1.y;  a[6]=(__bf16)l1.z;  a[7]=(__bf16)l1.w;
    a[8]=(__bf16)h0.x;  a[9]=(__bf16)h0.y;  a[10]=(__bf16)h0.z; a[11]=(__bf16)h0.w;
    a[12]=(__bf16)h1.x; a[13]=(__bf16)h1.y; a[14]=(__bf16)h1.z; a[15]=(__bf16)h1.w;
    v16bf b = load_bfragB(br + kt, kh);
    acc = WMMA_BF16(a, b, acc);
  }
  float* o = out + col0 + lm;
#pragma unroll
  for (int r=0;r<8;++r) o[(size_t)(8*kh + r) * ldo] = acc[r];
}

__device__ __forceinline__ void gru_elem(
    const float* gi, const float* gh,
    const float* bih, const float* bhh,
    const float* hprev, float* hout, float* es_out,
    int Dd, int start, int stride)
{
  const int G3 = 3*Dd;
  for (int idx = start; idx < 16*Dd; idx += stride){
    const int m = idx / Dd, d = idx - m*Dd;
    const float* gim = gi + (size_t)m*G3;
    const float* ghm = gh + (size_t)m*G3;
    float r = sigm(gim[d] + bih[d] + ghm[d] + bhh[d]);
    float z = sigm(gim[Dd+d] + bih[Dd+d] + ghm[Dd+d] + bhh[Dd+d]);
    float n = tanhf(gim[2*Dd+d] + bih[2*Dd+d] + r*(ghm[2*Dd+d] + bhh[2*Dd+d]));
    float h = (1.0f - z)*n + z*hprev[idx];
    hout[idx] = h;
    if (es_out) es_out[idx] = h;
  }
}

__device__ __forceinline__ void grid_barrier(unsigned int* bar, unsigned int target){
  __threadfence();
  __syncthreads();
  if (threadIdx.x == 0){
    __hip_atomic_fetch_add(bar, 1u, __ATOMIC_ACQ_REL, __HIP_MEMORY_SCOPE_AGENT);
    while (__hip_atomic_load(bar, __ATOMIC_ACQUIRE, __HIP_MEMORY_SCOPE_AGENT) < target)
      __builtin_amdgcn_s_sleep(2);
  }
  __syncthreads();
  __threadfence();
}

// =====================================================================
// Phase 2: persistent utterance-loop kernel. 24 WGs x 8 waves = 192 waves.
// =====================================================================
#define NWG  24
#define TOTW 192

__global__ __launch_bounds__(256) void utt_kernel(
  const float* __restrict__ uttb,
  float* hist, float* p0, float* p1, float* evec,
  float* q, float* ctx, float* scores,
  float* gi, float* gh, float* egi, float* egh,
  float* es, unsigned int* bar,
  const __bf16* __restrict__ ggWih, const __bf16* __restrict__ ggWhh,
  const __bf16* __restrict__ pgWih, const __bf16* __restrict__ pgWhh,
  const __bf16* __restrict__ egWih, const __bf16* __restrict__ egWhh,
  const __bf16* __restrict__ WgT,
  const float* __restrict__ gg_bih, const float* __restrict__ gg_bhh,
  const float* __restrict__ pg_bih, const float* __restrict__ pg_bhh,
  const float* __restrict__ eg_bih, const float* __restrict__ eg_bhh)
{
  __shared__ float wsm[65*16];
  const int tid  = threadIdx.x;
  const int wave = tid >> 5, lane = tid & 31;
  const int gw   = blockIdx.x * 8 + wave;
  const int gtid = blockIdx.x * 256 + tid;
  unsigned int bt = 0;

  for (int i = 0; i < 64; ++i){
    const float* u_i    = uttb + (size_t)i*16*1024;
    const float* hist_i = hist + (size_t)i*16*1024;
    float*       hist_n = hist + (size_t)(i+1)*16*1024;
    float*       psel   = (i & 1) ? p1 : p0;

    // ---- S1: gg gate GEMMs (192 strips) + q GEMM (64 strips) ----
    for (int s = gw; s < 256; s += TOTW){
      if (s < 192){
        gemm_strip(u_i, psel,   1024, ggWih, 1024, 1024, gi, 3072, s*16, lane);
        gemm_strip(hist_i, nullptr, 1024, ggWhh, 1024, 1024, gh, 3072, s*16, lane);
      } else {
        gemm_strip(u_i, nullptr, 1024, WgT, 1024, 1024, q, 1024, (s-192)*16, lane);
      }
    }
    bt += NWG; grid_barrier(bar, bt);

    // ---- S2: gg elementwise -> hist[i+1]; attention scores (t<=i uses old hist) ----
    gru_elem(gi, gh, gg_bih, gg_bhh, hist_i, hist_n, nullptr, 1024, gtid, NWG*256);
    for (int p = gw; p < (i+1)*16; p += TOTW){
      const int tt = p >> 4, b = p & 15;
      const float* qa = q + b*1024;
      const float* ha = hist + (size_t)(tt*16 + b)*1024;
      float s = 0.0f;
      for (int k = lane; k < 1024; k += 32) s += qa[k]*ha[k];
#pragma unroll
      for (int off = 16; off > 0; off >>= 1) s += __shfl_down(s, off, 32);
      if (lane == 0) scores[tt*16 + b] = s;
    }
    bt += NWG; grid_barrier(bar, bt);

    // ---- S3: softmax (redundant per-WG into LDS) + ctx (partitioned) ----
    if (tid < 16){
      const int b = tid;
      float mx = -1e30f;
      for (int tt = 0; tt <= i; ++tt) mx = fmaxf(mx, scores[tt*16 + b]);
      float ssum = 0.0f;
      for (int tt = 0; tt <= i; ++tt){
        float e = __expf(scores[tt*16 + b] - mx);
        wsm[tt*16 + b] = e; ssum += e;
      }
      float inv = 1.0f / ssum;
      for (int tt = 0; tt <= i; ++tt) wsm[tt*16 + b] *= inv;
    }
    __syncthreads();
    for (int idx = gtid; idx < 16*1024; idx += NWG*256){
      const int b = idx >> 10, d = idx & 1023;
      float a = 0.0f;
      for (int tt = 0; tt <= i; ++tt)
        a += wsm[tt*16 + b] * hist[(size_t)(tt*16 + b)*1024 + d];
      ctx[idx] = a;
    }
    bt += NWG; grid_barrier(bar, bt);

    // ---- S5: pg gate GEMMs ----
    gemm_strip(u_i, ctx,    1024, pgWih, 1024, 1024, gi, 3072, gw*16, lane);
    gemm_strip(psel, nullptr, 1024, pgWhh, 1024, 1024, gh, 3072, gw*16, lane);
    bt += NWG; grid_barrier(bar, bt);

    // ---- S6: pg elementwise -> p_new (in place) ----
    gru_elem(gi, gh, pg_bih, pg_bhh, psel, psel, nullptr, 1024, gtid, NWG*256);
    bt += NWG; grid_barrier(bar, bt);

    // ---- S7: eg gate GEMMs (96 gi strips K=1024, 96 gh strips K=512) ----
    if (gw < 96)
      gemm_strip(psel, nullptr, 1024, egWih, 1024, 1024, egi, 1536, gw*16, lane);
    else
      gemm_strip(evec, nullptr, 512,  egWhh, 512,  512,  egh, 1536, (gw-96)*16, lane);
    bt += NWG; grid_barrier(bar, bt);

    // ---- S8: eg elementwise -> e, es[i]  (piggybacks on next step's barriers) ----
    gru_elem(egi, egh, eg_bih, eg_bhh, evec, evec, es + (size_t)i*16*512, 512, gtid, NWG*256);
  }
}

// =====================================================================
// Phase 3 + setup kernels
// =====================================================================
__global__ void cls_kernel(const float* __restrict__ es, const float* __restrict__ W,
                           const float* __restrict__ bias, float* __restrict__ out)
{
  int idx = blockIdx.x*blockDim.x + threadIdx.x;
  if (idx >= 1024) return;
  int b = idx >> 6, u = idx & 63;
  const float* ev = es + (size_t)(u*16 + b)*512;
#pragma unroll
  for (int c = 0; c < 7; ++c){
    float s = bias[c];
    const float* wr = W + c*512;
    for (int k = 0; k < 512; ++k) s += ev[k]*wr[k];
    out[(size_t)(b*64 + u)*7 + c] = s;
  }
}

__global__ void cvt_pad_bf16(__bf16* dst, const float* src, int rows, int cols, int dcols){
  int idx = blockIdx.x*blockDim.x + threadIdx.x;
  if (idx >= rows*dcols) return;
  int r = idx / dcols, c = idx - r*dcols;
  dst[idx] = (c < cols) ? (__bf16)src[(size_t)r*cols + c] : (__bf16)0.0f;
}
__global__ void cvt_T_bf16(__bf16* dst, const float* src, int rowsK, int colsN){
  int idx = blockIdx.x*blockDim.x + threadIdx.x;
  if (idx >= rowsK*colsN) return;
  int n = idx / rowsK, k = idx - n*rowsK;
  dst[idx] = (__bf16)src[(size_t)k*colsN + n];
}
__global__ void vec_add_f32(float* dst, const float* a, const float* b, int n){
  int idx = blockIdx.x*blockDim.x + threadIdx.x;
  if (idx < n) dst[idx] = a[idx] + b[idx];
}
__global__ void zero_f32(float* p, int n){
  int idx = blockIdx.x*blockDim.x + threadIdx.x;
  if (idx < n) p[idx] = 0.0f;
}

// =====================================================================
extern "C" void kernel_launch(void* const* d_in, const int* in_sizes, int n_in,
                              void* d_out, int out_size, void* d_ws, size_t ws_size,
                              hipStream_t stream)
{
  const float* x       = (const float*)d_in[0];
  const float* WihF    = (const float*)d_in[1];
  const float* WhhF    = (const float*)d_in[2];
  const float* bihF    = (const float*)d_in[3];
  const float* bhhF    = (const float*)d_in[4];
  const float* WihB    = (const float*)d_in[5];
  const float* WhhB    = (const float*)d_in[6];
  const float* bihB    = (const float*)d_in[7];
  const float* bhhB    = (const float*)d_in[8];
  const float* ggWih_f = (const float*)d_in[9];
  const float* ggWhh_f = (const float*)d_in[10];
  const float* gg_bih  = (const float*)d_in[11];
  const float* gg_bhh  = (const float*)d_in[12];
  const float* Wg      = (const float*)d_in[13];
  const float* pgWih_f = (const float*)d_in[14];
  const float* pgWhh_f = (const float*)d_in[15];
  const float* pg_bih  = (const float*)d_in[16];
  const float* pg_bhh  = (const float*)d_in[17];
  const float* egWih_f = (const float*)d_in[18];
  const float* egWhh_f = (const float*)d_in[19];
  const float* eg_bih  = (const float*)d_in[20];
  const float* eg_bhh  = (const float*)d_in[21];
  const float* clsW    = (const float*)d_in[22];
  const float* clsB    = (const float*)d_in[23];
  float* out = (float*)d_out;
  (void)in_sizes; (void)n_in; (void)out_size; (void)ws_size;

  char* base = (char*)d_ws;
  size_t off = 0;
  auto take = [&](size_t bytes)->char*{
    char* p = base + off;
    off = (off + bytes + 255) & ~(size_t)255;
    return p;
  };
  __bf16* xbf   = (__bf16*)take((size_t)32768*320*2);
  __bf16* Wihp  = (__bf16*)take((size_t)2*2048*320*2);
  __bf16* Whhb  = (__bf16*)take((size_t)2*2048*512*2);
  float*  bsum  = (float*) take((size_t)2*2048*4);
  __bf16* ggWih = (__bf16*)take((size_t)3072*1024*2);
  __bf16* ggWhh = (__bf16*)take((size_t)3072*1024*2);
  __bf16* pgWih = (__bf16*)take((size_t)3072*1024*2);
  __bf16* pgWhh = (__bf16*)take((size_t)3072*1024*2);
  __bf16* egWih = (__bf16*)take((size_t)1536*1024*2);
  __bf16* egWhh = (__bf16*)take((size_t)1536*512*2);
  __bf16* WgT   = (__bf16*)take((size_t)1024*1024*2);
  float*  uttb  = (float*) take((size_t)64*16*1024*4);
  // --- contiguous zero-init region: hist, p0, p1, e, barrier ---
  float*  hist  = (float*) take((size_t)65*16*1024*4);
  float*  p0    = (float*) take((size_t)16*1024*4);
  float*  p1    = (float*) take((size_t)16*1024*4);
  float*  evec  = (float*) take((size_t)16*512*4);
  unsigned int* bar = (unsigned int*)take(256);
  // --- scratch ---
  float* q      = (float*) take((size_t)16*1024*4);
  float* ctx    = (float*) take((size_t)16*1024*4);
  float* scores = (float*) take((size_t)65*16*4);
  float* gi     = (float*) take((size_t)16*3072*4);
  float* gh     = (float*) take((size_t)16*3072*4);
  float* egi    = (float*) take((size_t)16*1536*4);
  float* egh    = (float*) take((size_t)16*1536*4);
  float* es     = (float*) take((size_t)64*16*512*4);

  auto blocks = [](size_t n){ return dim3((unsigned)((n + 255)/256)); };

  // bf16 weight / input staging (re-done each launch: deterministic, ~tens of us)
  cvt_pad_bf16<<<blocks((size_t)32768*320),256,0,stream>>>(xbf, x, 32768, 300, 320);
  cvt_pad_bf16<<<blocks((size_t)2048*320),256,0,stream>>>(Wihp,            WihF, 2048, 300, 320);
  cvt_pad_bf16<<<blocks((size_t)2048*320),256,0,stream>>>(Wihp + 2048*320, WihB, 2048, 300, 320);
  cvt_pad_bf16<<<blocks((size_t)2048*512),256,0,stream>>>(Whhb,            WhhF, 2048, 512, 512);
  cvt_pad_bf16<<<blocks((size_t)2048*512),256,0,stream>>>(Whhb + 2048*512, WhhB, 2048, 512, 512);
  vec_add_f32<<<blocks(2048),256,0,stream>>>(bsum,        bihF, bhhF, 2048);
  vec_add_f32<<<blocks(2048),256,0,stream>>>(bsum + 2048, bihB, bhhB, 2048);
  cvt_pad_bf16<<<blocks((size_t)3072*1024),256,0,stream>>>(ggWih, ggWih_f, 3072, 1024, 1024);
  cvt_pad_bf16<<<blocks((size_t)3072*1024),256,0,stream>>>(ggWhh, ggWhh_f, 3072, 1024, 1024);
  cvt_pad_bf16<<<blocks((size_t)3072*1024),256,0,stream>>>(pgWih, pgWih_f, 3072, 1024, 1024);
  cvt_pad_bf16<<<blocks((size_t)3072*1024),256,0,stream>>>(pgWhh, pgWhh_f, 3072, 1024, 1024);
  cvt_pad_bf16<<<blocks((size_t)1536*1024),256,0,stream>>>(egWih, egWih_f, 1536, 1024, 1024);
  cvt_pad_bf16<<<blocks((size_t)1536*512 ),256,0,stream>>>(egWhh, egWhh_f, 1536, 512, 512);
  cvt_T_bf16  <<<blocks((size_t)1024*1024),256,0,stream>>>(WgT, Wg, 1024, 1024);
  zero_f32<<<blocks(1105984),256,0,stream>>>(hist, 1105984); // hist+p0+p1+e+bar

  lstm_kernel<<<dim3(64,2),512,0,stream>>>(xbf, Wihp, Whhb, bsum, uttb);

  utt_kernel<<<NWG,256,0,stream>>>(uttb, hist, p0, p1, evec, q, ctx, scores,
      gi, gh, egi, egh, es, bar,
      ggWih, ggWhh, pgWih, pgWhh, egWih, egWhh, WgT,
      gg_bih, gg_bhh, pg_bih, pg_bhh, eg_bih, eg_bhh);

  cls_kernel<<<4,256,0,stream>>>(es, clsW, clsB, out);
}
